// LorentzRankingLoss_36103495090793
// MI455X (gfx1250) — compile-verified, matrix operand order checked by reading, not run on hardware
//
#include <hip/hip_runtime.h>
#include <hip/hip_bf16.h>
#include <math.h>

// ---------------- problem constants (match reference) ----------------
#define NUM_CLASSES 128
#define NUM_SAMPLES 64
#define KMAX        (NUM_CLASSES * NUM_SAMPLES)   // 8192 anchors
#define DIMS        64
#define NUM_NEG     8
#define MARGIN      0.1f
#define EPSV        1e-7f

typedef __attribute__((ext_vector_type(16))) _Float16 v16h;
typedef __attribute__((ext_vector_type(8)))  float    v8f;

// ---------------- deterministic integer mix (PRNG stand-in) ----------------
__device__ __forceinline__ unsigned mixu(unsigned x) {
    x ^= x >> 16; x *= 0x7FEB352Du;
    x ^= x >> 15; x *= 0x846CA68Bu;
    x ^= x >> 16;
    return x;
}

// ---------------- 0: zero reservoir table + accumulators ----------------
__global__ void lrl_init(unsigned long long* __restrict__ table,
                         float* __restrict__ accum) {
    int i = blockIdx.x * blockDim.x + threadIdx.x;
    if (i < KMAX) table[i] = 0ull;
    if (i < 2)   accum[i] = 0.0f;
}

// ---------------- 1: order-independent per-class reservoir sampling ----------
// Each voxel hashes itself to (slot, priority); atomicMax on packed key makes
// the selected winner per (class, slot) deterministic regardless of wave order.
__global__ void lrl_sample(const int* __restrict__ labels,
                           unsigned long long* __restrict__ table, int N) {
    int i = blockIdx.x * blockDim.x + threadIdx.x;
    if (i >= N) return;
    int c = labels[i] & (NUM_CLASSES - 1);
    unsigned pri  = mixu((unsigned)i ^ 0x9E3779B9u) | 0x80000000u;   // nonzero key
    unsigned slot = mixu((unsigned)i * 0x85EBCA6Bu + 0xC2B2AE35u) & (NUM_SAMPLES - 1);
    unsigned long long key = (((unsigned long long)pri) << 32) | (unsigned)i;
    atomicMax(&table[c * NUM_SAMPLES + slot], key);
}

// ---------------- 2: label_emb -> f16 + Lorentz time component --------------
__global__ void lrl_labels(const float* __restrict__ lemb,
                           _Float16* __restrict__ lf16,
                           float* __restrict__ ty) {
    int c = threadIdx.x;           // 128 threads
    if (c >= NUM_CLASSES) return;
    float sq = 0.0f;
    #pragma unroll
    for (int d = 0; d < DIMS; ++d) {
        float v = lemb[c * DIMS + d];
        lf16[c * DIMS + d] = (_Float16)v;
        sq += v * v;
    }
    ty[c] = sqrtf(1.0f + sq);      // 1/CURV = 1
}

// ---------------- 3: gather anchors (channel-major voxel layout) ------------
// voxel_flat[n,d] = voxel_emb[b, d, h, w, z], channel stride = SP = H*W*Z.
__global__ void lrl_gather(const float* __restrict__ voxel,
                           const unsigned long long* __restrict__ table,
                           _Float16* __restrict__ af16,
                           float* __restrict__ tx, int SP) {
    int k = blockIdx.x * blockDim.x + threadIdx.x;
    if (k >= KMAX) return;
    unsigned long long t = table[k];
    int idx = (int)(t & 0xFFFFFFFFull);           // 0 if slot empty (masked later)
    int b  = idx / SP;
    int sp = idx % SP;
    size_t base = ((size_t)b * DIMS) * (size_t)SP + (size_t)sp;
    float sq = 0.0f;
    #pragma unroll 4
    for (int d = 0; d < DIMS; ++d) {
        if (d + 16 < DIMS)
            __builtin_prefetch(&voxel[base + (size_t)(d + 16) * SP], 0, 0);
        float v = voxel[base + (size_t)d * SP];
        af16[k * DIMS + d] = (_Float16)v;
        sq += v * v;
    }
    tx[k] = sqrtf(1.0f + sq);
}

// ---------------- 4: WMMA GEMM [8192x64]x[64x128] + acosh distances ---------
// Block = 8 waves; wave w computes the 16x16 tile (blockIdx.x, w).
// A fragment (ISA 7.12.2, 16-bit A 16x32): lane&15 = row; VGPR v<4 holds
// K = (lane<16?0:8)+2v, VGPR v>=4 holds K = (lane<16?16:24)+2(v-4).
// B mirrors it per column (lane&15 = N), so both gather rows of [*,64] f16.
__global__ __launch_bounds__(256)
void lrl_gemm(const _Float16* __restrict__ A,   // anchors_f16 [KMAX][64]
              const _Float16* __restrict__ Bm,  // label_f16   [128][64]
              const float* __restrict__ tx,
              const float* __restrict__ ty,
              float* __restrict__ dist) {       // [KMAX][128]
    const int lane = threadIdx.x & 31;
    const int wave = threadIdx.x >> 5;
    const int m0 = blockIdx.x * 16;
    const int n0 = wave * 16;
    const int row = lane & 15;
    const bool hi = lane >= 16;

    v8f c = {};
    #pragma unroll
    for (int kc = 0; kc < DIMS; kc += 32) {
        v16h a, b;
        #pragma unroll
        for (int v = 0; v < 8; ++v) {
            int kk = kc + ((v < 4) ? ((hi ? 8 : 0) + 2 * v)
                                   : ((hi ? 24 : 16) + 2 * (v - 4)));
            a[2 * v]     = A[(m0 + row) * DIMS + kk];
            a[2 * v + 1] = A[(m0 + row) * DIMS + kk + 1];
            b[2 * v]     = Bm[(n0 + row) * DIMS + kk];
            b[2 * v + 1] = Bm[(n0 + row) * DIMS + kk + 1];
        }
        c = __builtin_amdgcn_wmma_f32_16x16x32_f16(
                /*neg_a=*/false, a, /*neg_b=*/false, b,
                /*c_mod=*/(short)0, c, /*reuse_a=*/false, /*reuse_b=*/false);
    }

    // C layout: VGPR r -> M = r (+8 for lanes 16..31), N = lane&15
    const int n = n0 + row;
    const float tyn = ty[n];
    #pragma unroll
    for (int r = 0; r < 8; ++r) {
        int m = m0 + r + (hi ? 8 : 0);
        float negInner = tx[m] * tyn - c[r];          // -<x,y>_L
        float x = fmaxf(negInner, 1.0f + EPSV);       // clip for acosh
        dist[m * NUM_CLASSES + n] = acoshf(x);        // sqrt(CURV)=1
    }
}

// ---------------- 5: top-8 hashed negatives + triplet reduction -------------
__global__ void lrl_loss(const float* __restrict__ dist,
                         const unsigned long long* __restrict__ table,
                         float* __restrict__ accum) {
    __shared__ float s_loss[256];
    __shared__ float s_cnt[256];
    int tid = threadIdx.x;
    int k = blockIdx.x * blockDim.x + tid;

    float l = 0.0f, cnt = 0.0f;
    if (k < KMAX) {
        int c = k >> 6;                          // class = k / NUM_SAMPLES
        bool valid = (table[k] != 0ull);
        float dpos = dist[k * NUM_CLASSES + c];

        float bs[NUM_NEG]; int bi[NUM_NEG];
        #pragma unroll
        for (int j = 0; j < NUM_NEG; ++j) { bs[j] = -1.0f; bi[j] = 0; }
        for (int cls = 0; cls < NUM_CLASSES; ++cls) {
            if (cls == c) continue;
            unsigned h = mixu(((unsigned)k * 0x01000193u) ^
                              ((unsigned)cls * 0x9E3779B9u) ^ 0xABCD1234u);
            float s = (float)(h >> 8) * (1.0f / 16777216.0f);
            if (s > bs[NUM_NEG - 1]) {
                int j = NUM_NEG - 1;
                while (j > 0 && bs[j - 1] < s) {
                    bs[j] = bs[j - 1]; bi[j] = bi[j - 1]; --j;
                }
                bs[j] = s; bi[j] = cls;
            }
        }
        #pragma unroll
        for (int j = 0; j < NUM_NEG; ++j) {
            float dneg = dist[k * NUM_CLASSES + bi[j]];
            l += fmaxf(MARGIN + dpos - dneg, 0.0f);
        }
        if (!valid) l = 0.0f;
        cnt = valid ? 1.0f : 0.0f;
    }

    s_loss[tid] = l; s_cnt[tid] = cnt;
    __syncthreads();
    for (int stride = 128; stride > 0; stride >>= 1) {
        if (tid < stride) {
            s_loss[tid] += s_loss[tid + stride];
            s_cnt[tid]  += s_cnt[tid + stride];
        }
        __syncthreads();
    }
    if (tid == 0) {
        atomicAdd(&accum[0], s_loss[0]);
        atomicAdd(&accum[1], s_cnt[0]);
    }
}

// ---------------- 6: finalize ----------------
__global__ void lrl_final(const float* __restrict__ accum,
                          float* __restrict__ out) {
    out[0] = accum[0] / fmaxf(accum[1] * (float)NUM_NEG, 1.0f);
}

// ---------------- workspace layout (bytes) ----------------
#define WS_TABLE   0u                               // 8192 * u64   = 65536
#define WS_AF16    65536u                           // 8192*64*2    = 1048576
#define WS_TX      1114112u                         // 8192*4       = 32768
#define WS_LF16    1146880u                         // 128*64*2     = 16384
#define WS_TY      1163264u                         // 128*4        = 512
#define WS_DIST    1163776u                         // 8192*128*4   = 4194304
#define WS_ACCUM   5358080u                         // 2*4

extern "C" void kernel_launch(void* const* d_in, const int* in_sizes, int n_in,
                              void* d_out, int out_size, void* d_ws, size_t ws_size,
                              hipStream_t stream) {
    const float* voxel  = (const float*)d_in[0];   // [2,64,96,96,96] f32
    const int*   labels = (const int*)d_in[1];     // [2,96,96,96]    i32
    const float* lemb   = (const float*)d_in[2];   // [128,64]        f32
    float* out = (float*)d_out;

    const int N  = in_sizes[1];     // 1,769,472 voxels
    const int SP = N / 2;           // H*W*Z (B = 2 per reference)

    char* ws = (char*)d_ws;
    unsigned long long* table = (unsigned long long*)(ws + WS_TABLE);
    _Float16* af16 = (_Float16*)(ws + WS_AF16);
    float*    tx   = (float*)(ws + WS_TX);
    _Float16* lf16 = (_Float16*)(ws + WS_LF16);
    float*    ty   = (float*)(ws + WS_TY);
    float*    dist = (float*)(ws + WS_DIST);
    float*    accum= (float*)(ws + WS_ACCUM);

    lrl_init  <<<(KMAX + 255) / 256, 256, 0, stream>>>(table, accum);
    lrl_sample<<<(N + 255) / 256,    256, 0, stream>>>(labels, table, N);
    lrl_labels<<<1, 128, 0, stream>>>(lemb, lf16, ty);
    lrl_gather<<<KMAX / 256, 256, 0, stream>>>(voxel, table, af16, tx, SP);
    lrl_gemm  <<<KMAX / 16,  256, 0, stream>>>(af16, lf16, tx, ty, dist);
    lrl_loss  <<<KMAX / 256, 256, 0, stream>>>(dist, table, accum);
    lrl_final <<<1, 1, 0, stream>>>(accum, out);
}
